// MultiHeadAttention_72318659330651
// MI455X (gfx1250) — compile-verified, matrix operand order checked by reading, not run on hardware
//
#include <hip/hip_runtime.h>
#include <hip/hip_bf16.h>

// MHA: N,H,W,L,C = 2,16,16,256,256 ; HEADS=8, hd=32 ; NHW = B = 512 blocks.
// 3-kernel bf16-WMMA pipeline:
//   1) qkv_kernel : x(f32) @ {Wq,Wk,Wv} -> q,k,v bf16  [b][head][L][hd]
//   2) attn_kernel: flash attention per (b,head)       -> attn bf16 [b][L][C]
//   3) proj_kernel: attn @ Wp + bp -> out f32
// Workspace: 4 x 64MB bf16 = 256 MB.

typedef __bf16 bf16;
typedef __attribute__((ext_vector_type(16))) __bf16 v16bf;
typedef __attribute__((ext_vector_type(8)))  __bf16 v8bf;
typedef __attribute__((ext_vector_type(8)))  float  v8f;

#define NB      512            // N*H*W
#define LSEQ    256
#define CCH     256
#define NHEADS  8
#define HD      32
#define QKV_ELEMS ((size_t)NB * NHEADS * LSEQ * HD)   // 33,554,432

__device__ __forceinline__ v8f wmma_bf16(v16bf a, v16bf b, v8f c) {
    return __builtin_amdgcn_wmma_f32_16x16x32_bf16(
        /*neg_a=*/false, a, /*neg_b=*/false, b,
        /*c_mod=*/(short)0, c, /*reuse_a=*/false, /*reuse_b=*/false);
}

// Load a 16-element bf16 fragment slice for one lane.
// A/B 16x32 bf16 layout: element e -> K = 16*(e>>3) + 8*g + (e&7)
//   => elements 0..7  at base + 8*g        (16 bytes, ds_load_b128)
//      elements 8..15 at base + 16 + 8*g   (16 bytes, ds_load_b128)
__device__ __forceinline__ v16bf frag16(const bf16* base, int g) {
    v8bf lo = *(const v8bf*)(base + g * 8);
    v8bf hi = *(const v8bf*)(base + 16 + g * 8);
    v16bf r;
#pragma unroll
    for (int i = 0; i < 8; ++i) { r[i] = lo[i]; r[i + 8] = hi[i]; }
    return r;
}

// A-fragment from fp32 row-major source (row pointer already offset to K-block).
__device__ __forceinline__ v16bf a_from_f32(const float* rp, int g) {
    float4 x0 = *(const float4*)(rp + g * 8);
    float4 x1 = *(const float4*)(rp + g * 8 + 4);
    float4 x2 = *(const float4*)(rp + 16 + g * 8);
    float4 x3 = *(const float4*)(rp + 16 + g * 8 + 4);
    v16bf a;
    a[0]  = (bf16)x0.x; a[1]  = (bf16)x0.y; a[2]  = (bf16)x0.z; a[3]  = (bf16)x0.w;
    a[4]  = (bf16)x1.x; a[5]  = (bf16)x1.y; a[6]  = (bf16)x1.z; a[7]  = (bf16)x1.w;
    a[8]  = (bf16)x2.x; a[9]  = (bf16)x2.y; a[10] = (bf16)x2.z; a[11] = (bf16)x2.w;
    a[12] = (bf16)x3.x; a[13] = (bf16)x3.y; a[14] = (bf16)x3.z; a[15] = (bf16)x3.w;
    return a;
}

// Stage W (256x256 f32, [k][n]) transposed+bf16 into WT[n][k] (padded rows).
// float4 global loads; adjacent-k bf16 pairs merge into ds_store_b32.
__device__ __forceinline__ void stage_w(const float* __restrict__ W,
                                        bf16 (*WT)[264], int tid) {
    for (int idx = tid; idx < 8192; idx += 256) {     // 8192 groups of (2k x 4n)
        int k0 = (idx >> 6) * 2, n0 = (idx & 63) * 4;
        float4 a0 = *(const float4*)(W + (size_t)k0 * 256 + n0);
        float4 a1 = *(const float4*)(W + (size_t)(k0 + 1) * 256 + n0);
        WT[n0 + 0][k0] = (bf16)a0.x; WT[n0 + 0][k0 + 1] = (bf16)a1.x;
        WT[n0 + 1][k0] = (bf16)a0.y; WT[n0 + 1][k0 + 1] = (bf16)a1.y;
        WT[n0 + 2][k0] = (bf16)a0.z; WT[n0 + 2][k0 + 1] = (bf16)a1.z;
        WT[n0 + 3][k0] = (bf16)a0.w; WT[n0 + 3][k0 + 1] = (bf16)a1.w;
    }
}

// ---------------- Kernel 1: QKV projection -----------------------------------
// grid = (2048, 3), block = 256 (8 waves). Block computes 64 rows x 256 cols
// of one of q/k/v. W staged transposed+bf16 in LDS once per block.
__global__ void qkv_kernel(const float* __restrict__ x,
                           const float* __restrict__ Wq, const float* __restrict__ bq,
                           const float* __restrict__ Wk, const float* __restrict__ bk,
                           const float* __restrict__ Wv, const float* __restrict__ bv,
                           bf16* __restrict__ qo, bf16* __restrict__ ko, bf16* __restrict__ vo) {
    __shared__ bf16 WT[256][264];   // [col n][k], 528B rows (16B multiple)

    const int mat = blockIdx.y;
    const float* W    = (mat == 0) ? Wq : (mat == 1) ? Wk : Wv;
    const float* bias = (mat == 0) ? bq : (mat == 1) ? bk : bv;
    bf16* out         = (mat == 0) ? qo : (mat == 1) ? ko : vo;

    const int tid = threadIdx.x;
    stage_w(W, WT, tid);
    __syncthreads();

    const int wave = tid >> 5, lane = tid & 31;
    const int g = lane >> 4, n16 = lane & 15;
    const int row0 = blockIdx.x * 64 + (wave >> 1) * 16;
    const int col0 = (wave & 1) * 128;

    v16bf af[8];                                   // 16 rows x full K=256
#pragma unroll
    for (int kb = 0; kb < 8; ++kb)
        af[kb] = a_from_f32(x + (size_t)(row0 + n16) * CCH + kb * 32, g);

#pragma unroll
    for (int ct = 0; ct < 8; ++ct) {
        const int col = col0 + ct * 16;
        v8f acc = {};
        v16bf bcur = frag16(&WT[col + n16][0], g);     // pipelined B frags
#pragma unroll
        for (int kb = 0; kb < 8; ++kb) {
            v16bf bnxt = bcur;
            if (kb < 7) bnxt = frag16(&WT[col + n16][(kb + 1) * 32], g);
            acc = wmma_bf16(af[kb], bcur, acc);
            bcur = bnxt;
        }
        const int j = col + n16;
        const float bj = bias[j];
        const int ah = j >> 5, d = j & 31;
#pragma unroll
        for (int r = 0; r < 8; ++r) {
            const int R = row0 + r + 8 * g;         // flat row = b*L + l
            const int b = R >> 8, l = R & 255;
            out[((size_t)(b * NHEADS + ah) * LSEQ + l) * HD + d] = (bf16)(acc[r] + bj);
        }
    }
}

// ---------------- Kernel 2: flash attention per (b, head) --------------------
// grid = 4096, block = 256 (8 waves). Wave owns 32 query rows. K,V in LDS.
__global__ void attn_kernel(const bf16* __restrict__ q, const bf16* __restrict__ k,
                            const bf16* __restrict__ v, bf16* __restrict__ ao) {
    __shared__ bf16 Ksh[256][32];      // [key l][d]        16 KB
    __shared__ bf16 VTsh[32][264];     // [d][key l] padded 16.5 KB
    __shared__ bf16 Psh[8][16][40];    // per-wave P transpose scratch (80B rows)

    const int bh = blockIdx.x;
    const int b = bh >> 3, ah = bh & 7;
    const size_t base = (size_t)bh * LSEQ * HD;
    const int tid = threadIdx.x;

    {   // K: contiguous 16KB -> LDS via async copy (ASYNCcnt path), 16B/lane
        const char* gsrc = (const char*)(k + base);
        unsigned lds0 = (unsigned)(unsigned long long)(&Ksh[0][0]);
#pragma unroll
        for (int it = 0; it < 4; ++it) {
            int i = tid + it * 256;                    // 1024 chunks of 16B
            unsigned lofs = lds0 + (unsigned)i * 16u;
            const void* gp = gsrc + (size_t)i * 16;
            asm volatile("global_load_async_to_lds_b128 %0, %1, off"
                         :: "v"(lofs), "v"(gp) : "memory");
        }
    }
    for (int i = tid; i < 256 * 32; i += 256) {        // V transposed (manual)
        int l = i >> 5, d = i & 31;
        VTsh[d][l] = v[base + i];
    }
    asm volatile("s_wait_asynccnt 0" ::: "memory");    // K copy done
    __syncthreads();

    const int wave = tid >> 5, lane = tid & 31;
    const int g = lane >> 4, n16 = lane & 15;
    const int l0 = wave * 32;

    v16bf qa[2];
#pragma unroll
    for (int rt = 0; rt < 2; ++rt)
        qa[rt] = frag16(q + base + (size_t)(l0 + rt * 16 + n16) * HD, g);

    v8f o[2][2] = {};
    float mrun[2][8], lrun[2][8];
#pragma unroll
    for (int rt = 0; rt < 2; ++rt)
#pragma unroll
        for (int r = 0; r < 8; ++r) { mrun[rt][r] = -1e30f; lrun[rt][r] = 0.f; }

    const float scale = 0.1767766952966369f;       // 1/sqrt(32)

    for (int kb = 0; kb < 8; ++kb) {               // 32 keys per step
        // K/V fragments depend only on kb: load once, reuse for both row tiles.
        v16bf kf0 = frag16(&Ksh[kb * 32 + n16][0], g);
        v16bf kf1 = frag16(&Ksh[kb * 32 + 16 + n16][0], g);
        v16bf vb0 = frag16(&VTsh[n16][kb * 32], g);
        v16bf vb1 = frag16(&VTsh[16 + n16][kb * 32], g);
#pragma unroll
        for (int rt = 0; rt < 2; ++rt) {
            v8f s0 = {}, s1 = {};
            s0 = wmma_bf16(qa[rt], kf0, s0);
            s1 = wmma_bf16(qa[rt], kf1, s1);
#pragma unroll
            for (int r = 0; r < 8; ++r) {          // online softmax per row
                float v0 = s0[r] * scale, v1 = s1[r] * scale;
                float tmax = fmaxf(v0, v1);
#pragma unroll
                for (int m = 8; m >= 1; m >>= 1) tmax = fmaxf(tmax, __shfl_xor(tmax, m, 32));
                float mnew = fmaxf(mrun[rt][r], tmax);
                float f = __expf(mrun[rt][r] - mnew);
                mrun[rt][r] = mnew;
                float p0 = __expf(v0 - mnew), p1 = __expf(v1 - mnew);
                float psum = p0 + p1;
#pragma unroll
                for (int m = 8; m >= 1; m >>= 1) psum += __shfl_xor(psum, m, 32);
                lrun[rt][r] = lrun[rt][r] * f + psum;
                o[rt][0][r] *= f; o[rt][1][r] *= f;
                Psh[wave][r + 8 * g][n16]      = (bf16)p0;   // C-layout -> LDS
                Psh[wave][r + 8 * g][16 + n16] = (bf16)p1;
            }
            asm volatile("s_wait_dscnt 0" ::: "memory");     // P store->load, same wave
            v16bf pa = frag16(&Psh[wave][n16][0], g);        // re-read in A layout
            o[rt][0] = wmma_bf16(pa, vb0, o[rt][0]);
            o[rt][1] = wmma_bf16(pa, vb1, o[rt][1]);
        }
    }

#pragma unroll
    for (int rt = 0; rt < 2; ++rt)
#pragma unroll
        for (int t = 0; t < 2; ++t) {
            const int col = ah * HD + t * 16 + n16;          // heads -> C layout
#pragma unroll
            for (int r = 0; r < 8; ++r) {
                const int l = l0 + rt * 16 + r + 8 * g;
                ao[((size_t)b * LSEQ + l) * CCH + col] = (bf16)(o[rt][t][r] / lrun[rt][r]);
            }
        }
}

// ---------------- Kernel 3: output projection --------------------------------
// grid = 2048, block = 256. attn(bf16) @ Wp + bp -> out f32.
__global__ void proj_kernel(const bf16* __restrict__ attn, const float* __restrict__ Wp,
                            const float* __restrict__ bp, float* __restrict__ out) {
    __shared__ bf16 WT[256][264];
    const int tid = threadIdx.x;
    stage_w(Wp, WT, tid);
    __syncthreads();

    const int wave = tid >> 5, lane = tid & 31;
    const int g = lane >> 4, n16 = lane & 15;
    const int row0 = blockIdx.x * 64 + (wave >> 1) * 16;
    const int col0 = (wave & 1) * 128;

    v16bf af[8];
#pragma unroll
    for (int kb = 0; kb < 8; ++kb)
        af[kb] = frag16(attn + (size_t)(row0 + n16) * CCH + kb * 32, g);

#pragma unroll
    for (int ct = 0; ct < 8; ++ct) {
        const int col = col0 + ct * 16;
        v8f acc = {};
        v16bf bcur = frag16(&WT[col + n16][0], g);
#pragma unroll
        for (int kb = 0; kb < 8; ++kb) {
            v16bf bnxt = bcur;
            if (kb < 7) bnxt = frag16(&WT[col + n16][(kb + 1) * 32], g);
            acc = wmma_bf16(af[kb], bcur, acc);
            bcur = bnxt;
        }
        const float bj = bp[col + n16];
#pragma unroll
        for (int r = 0; r < 8; ++r) {
            const int R = row0 + r + 8 * g;
            out[(size_t)R * CCH + col + n16] = acc[r] + bj;
        }
    }
}

extern "C" void kernel_launch(void* const* d_in, const int* in_sizes, int n_in,
                              void* d_out, int out_size, void* d_ws, size_t ws_size,
                              hipStream_t stream) {
    const float* x  = (const float*)d_in[0];
    const float* Wq = (const float*)d_in[1]; const float* bq = (const float*)d_in[2];
    const float* Wk = (const float*)d_in[3]; const float* bk = (const float*)d_in[4];
    const float* Wv = (const float*)d_in[5]; const float* bv = (const float*)d_in[6];
    const float* Wp = (const float*)d_in[7]; const float* bp = (const float*)d_in[8];

    bf16* qws = (bf16*)d_ws;
    bf16* kws = qws + QKV_ELEMS;
    bf16* vws = kws + QKV_ELEMS;
    bf16* aws = vws + QKV_ELEMS;

    const int Mrows = NB * LSEQ;                 // 131072
    dim3 g1(Mrows / 64, 3);
    qkv_kernel<<<g1, 256, 0, stream>>>(x, Wq, bq, Wk, bk, Wv, bv, qws, kws, vws);
    attn_kernel<<<NB * NHEADS, 256, 0, stream>>>(qws, kws, vws, aws);
    proj_kernel<<<Mrows / 64, 256, 0, stream>>>(aws, Wp, bp, (float*)d_out);
}
